// HKGRippleNetModel_49512382988743
// MI455X (gfx1250) — matrix operand order repaired and checked by reading, not compiled, not run on hardware
//
#include <hip/hip_runtime.h>
#include <hip/hip_bf16.h>

// ---------------------------------------------------------------------------
// Types for CDNA5 WMMA (wave32): v_wmma_f32_16x16x32_bf16
// ---------------------------------------------------------------------------
typedef __bf16 bf16_t;
typedef bf16_t v16bf __attribute__((ext_vector_type(16)));
typedef float  v8f   __attribute__((ext_vector_type(8)));

#define LEAKY 0.1f

// Native f32 -> bf16 (RNE)
__device__ __forceinline__ bf16_t f2bf(float f) { return (bf16_t)f; }

// B-fragment-friendly LDS address for a 128x128 (k,n) matrix, tiled as
// (nt, s) 32x16 tiles; lane L holds column n=L&15, K split 0-7/16-23 (table order)
__device__ __forceinline__ int b_addr(int k, int n) {
    int s  = k >> 5, kk = k & 31, nt = n >> 4;
    int L  = (((kk >> 3) & 1) << 4) | (n & 15);
    int i  = (kk & 7) | ((kk >> 4) << 3);
    return (((nt << 2) + s) * 32 + L) * 16 + i;
}

// A-fragment address within one 16x32 tile: lane L row m=L&15, K split 0-7/16-23
__device__ __forceinline__ int a_tile_addr(int kk, int mloc) {
    int L = (((kk >> 3) & 1) << 4) | mloc;
    int i = (kk & 7) | ((kk >> 4) << 3);
    return L * 16 + i;
}

__device__ __forceinline__ float lrelu_f(float x) { return x > 0.f ? x : LEAKY * x; }

// ---------------------------------------------------------------------------
// out[N,128] = epilogue( A[N,128] @ W[128,128] ).  EP: 0 none, 1 lrelu, 2 relu
// 8 waves/block, 16 rows per wave, W staged once to LDS as bf16 fragments.
// Safe in-place (out == A): all A reads complete before any store.
// ---------------------------------------------------------------------------
template <int EP>
__global__ __launch_bounds__(256)
void gemm128_kernel(const float* __restrict__ A, const float* __restrict__ W,
                    float* __restrict__ out, int N) {
    __shared__ __align__(32) bf16_t wlds[128 * 128];
    const int tid = threadIdx.x;
    #pragma unroll 4
    for (int it = 0; it < 64; ++it) {
        int idx = it * 256 + tid;
        int k = idx >> 7, n = idx & 127;
        wlds[b_addr(k, n)] = f2bf(W[idx]);
    }
    __syncthreads();

    const int wave = tid >> 5, L = tid & 31;
    const long row0 = (long)blockIdx.x * 128 + wave * 16;

    v8f acc[8] = {};
    #pragma unroll
    for (int s = 0; s < 4; ++s) {
        const int  kb = s * 32 + ((L >> 4) << 3);
        const long r  = row0 + (L & 15);
        const float* p = A + r * 128 + kb;
        v16bf a;
        #pragma unroll
        for (int i = 0; i < 16; ++i) {
            float x = (r < N) ? p[i + ((i >= 8) ? 8 : 0)] : 0.0f;
            a[i] = f2bf(x);
        }
        #pragma unroll
        for (int j = 0; j < 8; ++j) {
            v16bf b = *(const v16bf*)&wlds[(((j << 2) + s) * 32 + L) * 16];
            acc[j] = __builtin_amdgcn_wmma_f32_16x16x32_bf16(
                false, a, false, b, (short)0, acc[j], false, false);
        }
    }

    const int mloc = (L >> 4) << 3;
    if (row0 + 16 <= (long)N) {              // fast path: full tile in range
        #pragma unroll
        for (int j = 0; j < 8; ++j) {
            #pragma unroll
            for (int rr = 0; rr < 8; ++rr) {
                float v = acc[j][rr];
                if (EP == 1) v = lrelu_f(v);
                else if (EP == 2) v = v > 0.f ? v : 0.f;
                out[(row0 + mloc + rr) * 128 + j * 16 + (L & 15)] = v;
            }
        }
    } else {
        #pragma unroll
        for (int j = 0; j < 8; ++j) {
            #pragma unroll
            for (int rr = 0; rr < 8; ++rr) {
                long m = row0 + mloc + rr;
                if (m < N) {
                    float v = acc[j][rr];
                    if (EP == 1) v = lrelu_f(v);
                    else if (EP == 2) v = v > 0.f ? v : 0.f;
                    out[m * 128 + j * 16 + (L & 15)] = v;
                }
            }
        }
    }
}

// ---------------------------------------------------------------------------
// acc128[128,128] += Hyp[N,128]^T @ Lat[N,128]   (split-K, RCHUNK rows/block)
// Double-buffered LDS + register prefetch: global loads for chunk k+1 are in
// flight while the WMMAs for chunk k execute.
// ---------------------------------------------------------------------------
#define RCHUNK 256
#define RSTEPS (RCHUNK / 32)
__global__ __launch_bounds__(256)
void hyperT_kernel(const float* __restrict__ Hyp, const float* __restrict__ Lat,
                   float* __restrict__ acc128, int N) {
    __shared__ __align__(32) bf16_t alds[2][32 * 128];
    __shared__ __align__(32) bf16_t blds[2][32 * 128];
    const int tid = threadIdx.x, wave = tid >> 5, L = tid & 31;
    const long n0 = (long)blockIdx.x * RCHUNK;

    // fixed per-thread staging coordinates: element (kk, c) for it = 0..15
    float hreg[16], lreg[16];

    // prefetch chunk 0 into registers, then store to buffer 0
    #pragma unroll
    for (int it = 0; it < 16; ++it) {
        int idx = it * 256 + tid;
        int kk = idx >> 7, c = idx & 127;
        long nr = n0 + kk;
        hreg[it] = (nr < N) ? Hyp[nr * 128 + c] : 0.f;
        lreg[it] = (nr < N) ? Lat[nr * 128 + c] : 0.f;
    }
    #pragma unroll
    for (int it = 0; it < 16; ++it) {
        int idx = it * 256 + tid;
        int kk = idx >> 7, c = idx & 127;
        alds[0][(c >> 4) * 512 + a_tile_addr(kk, c & 15)] = f2bf(hreg[it]);
        blds[0][((c >> 4) * 32 + ((((kk >> 3) & 1) << 4) | (c & 15))) * 16 + ((kk & 7) | ((kk >> 4) << 3))] = f2bf(lreg[it]);
    }
    __syncthreads();

    v8f acc[8] = {};
    for (int ks = 0; ks < RSTEPS; ++ks) {
        const int buf = ks & 1;
        // issue next chunk's global loads (overlap with WMMA below)
        if (ks + 1 < RSTEPS) {
            const long nb = n0 + (long)(ks + 1) * 32;
            #pragma unroll
            for (int it = 0; it < 16; ++it) {
                int idx = it * 256 + tid;
                int kk = idx >> 7, c = idx & 127;
                long nr = nb + kk;
                hreg[it] = (nr < N) ? Hyp[nr * 128 + c] : 0.f;
                lreg[it] = (nr < N) ? Lat[nr * 128 + c] : 0.f;
            }
        }
        // compute current chunk
        v16bf a = *(const v16bf*)&alds[buf][(wave * 32 + L) * 16];
        #pragma unroll
        for (int j = 0; j < 8; ++j) {
            v16bf b = *(const v16bf*)&blds[buf][(j * 32 + L) * 16];
            acc[j] = __builtin_amdgcn_wmma_f32_16x16x32_bf16(
                false, a, false, b, (short)0, acc[j], false, false);
        }
        __syncthreads();   // reads of buf^1 (prev iter) done before overwrite
        if (ks + 1 < RSTEPS) {
            #pragma unroll
            for (int it = 0; it < 16; ++it) {
                int idx = it * 256 + tid;
                int kk = idx >> 7, c = idx & 127;
                alds[buf ^ 1][(c >> 4) * 512 + a_tile_addr(kk, c & 15)] = f2bf(hreg[it]);
                blds[buf ^ 1][((c >> 4) * 32 + ((((kk >> 3) & 1) << 4) | (c & 15))) * 16 + ((kk & 7) | ((kk >> 4) << 3))] = f2bf(lreg[it]);
            }
            __syncthreads();
        }
    }
    const int mloc = (L >> 4) << 3;
    #pragma unroll
    for (int j = 0; j < 8; ++j)
        #pragma unroll
        for (int rr = 0; rr < 8; ++rr) {
            int h = wave * 16 + mloc + rr;
            int d = j * 16 + (L & 15);
            unsafeAtomicAdd(&acc128[h * 128 + d], acc[j][rr]);
        }
}

// ---------------------------------------------------------------------------
// lat = lrelu(lat); then 3x: lat = relu(W_t^T @ lat) + lat.  One block, WMMA.
// ---------------------------------------------------------------------------
__global__ __launch_bounds__(256)
void resblock_kernel(float* __restrict__ lat, const float* __restrict__ Ws) {
    __shared__ __align__(32) bf16_t alds[128 * 128];
    __shared__ __align__(32) bf16_t blds[128 * 128];
    const int tid = threadIdx.x, wave = tid >> 5, L = tid & 31;

    #pragma unroll 4
    for (int it = 0; it < 64; ++it) {
        int idx = it * 256 + tid;
        int k = idx >> 7, n = idx & 127;
        float v = lrelu_f(lat[idx]);
        lat[idx] = v;
        blds[b_addr(k, n)] = f2bf(v);
    }
    __syncthreads();

    for (int t = 0; t < 3; ++t) {
        const float* W = Ws + t * 16384;
        // stage A = W^T: A[m,k] = W[k,m]
        #pragma unroll 4
        for (int it = 0; it < 64; ++it) {
            int idx = it * 256 + tid;
            int k = idx >> 7, m = idx & 127;
            alds[(((m >> 4) << 2) + (k >> 5)) * 512 + a_tile_addr(k & 31, m & 15)] = f2bf(W[idx]);
        }
        __syncthreads();

        v8f acc[8] = {};
        #pragma unroll
        for (int s = 0; s < 4; ++s) {
            v16bf a = *(const v16bf*)&alds[(((wave << 2) + s) * 32 + L) * 16];
            #pragma unroll
            for (int j = 0; j < 8; ++j) {
                v16bf b = *(const v16bf*)&blds[(((j << 2) + s) * 32 + L) * 16];
                acc[j] = __builtin_amdgcn_wmma_f32_16x16x32_bf16(
                    false, a, false, b, (short)0, acc[j], false, false);
            }
        }
        __syncthreads();   // all LDS reads done before rewriting blds

        const int mloc = (L >> 4) << 3;
        #pragma unroll
        for (int j = 0; j < 8; ++j)
            #pragma unroll
            for (int rr = 0; rr < 8; ++rr) {
                int m = wave * 16 + mloc + rr;
                int n = j * 16 + (L & 15);
                float mm = acc[j][rr];
                float v = (mm > 0.f ? mm : 0.f) + lat[m * 128 + n];
                lat[m * 128 + n] = v;
                blds[b_addr(m, n)] = f2bf(v);   // lat is next iteration's B (k=m)
            }
        __syncthreads();
    }
}

// ---------------------------------------------------------------------------
// Edge-parallel SpMM: out[dst[e]] += vals[e] * src[srcIdx[e]]  (128-wide rows)
// one wave per edge, float4 per lane, f32 HW atomics
// ---------------------------------------------------------------------------
__global__ __launch_bounds__(256)
void spmm_kernel(const float* __restrict__ src, const int* __restrict__ dstIdx,
                 const int* __restrict__ srcIdx, const float* __restrict__ vals,
                 float* __restrict__ out, int E) {
    const int e = blockIdx.x * 8 + (threadIdx.x >> 5);
    if (e >= E) return;
    const int lane = threadIdx.x & 31;
    const long s = srcIdx[e], d = dstIdx[e];
    const float v = vals[e];
    const float4 x = *(const float4*)(src + s * 128 + lane * 4);
    float* o = out + d * 128 + lane * 4;
    unsafeAtomicAdd(o + 0, v * x.x);
    unsafeAtomicAdd(o + 1, v * x.y);
    unsafeAtomicAdd(o + 2, v * x.z);
    unsafeAtomicAdd(o + 3, v * x.w);
}

// ---------------------------------------------------------------------------
// Elementwise helpers (float4-vectorized; all sizes are multiples of 128)
// ---------------------------------------------------------------------------
__global__ __launch_bounds__(256)
void zero_kernel(float4* __restrict__ p, long n4) {
    long i = (long)blockIdx.x * 256 + threadIdx.x;
    if (i < n4) p[i] = make_float4(0.f, 0.f, 0.f, 0.f);
}
__global__ __launch_bounds__(256)
void copy2_kernel(const float4* __restrict__ s, float4* __restrict__ a,
                  float4* __restrict__ b, long n4) {
    long i = (long)blockIdx.x * 256 + threadIdx.x;
    if (i < n4) { float4 v = s[i]; a[i] = v; b[i] = v; }
}
__global__ __launch_bounds__(256)
void combine_kernel(const float4* __restrict__ g, const float4* __restrict__ h,
                    float4* __restrict__ lat, float4* __restrict__ sum, long n4) {
    long i = (long)blockIdx.x * 256 + threadIdx.x;
    if (i < n4) {
        float4 gv = g[i], hv = h[i], lv = lat[i], sv = sum[i];
        float4 v = make_float4(gv.x + hv.x + lv.x, gv.y + hv.y + lv.y,
                               gv.z + hv.z + lv.z, gv.w + hv.w + lv.w);
        lat[i] = v;
        sum[i] = make_float4(sv.x + v.x, sv.y + v.y, sv.z + v.z, sv.w + v.w);
    }
}

// ---------------------------------------------------------------------------
// Host orchestration
// ---------------------------------------------------------------------------
extern "C" void kernel_launch(void* const* d_in, const int* in_sizes, int n_in,
                              void* d_out, int out_size, void* d_ws, size_t ws_size,
                              hipStream_t stream) {
    const float* uEmbed0 = (const float*)d_in[0];
    const float* iEmbed0 = (const float*)d_in[1];
    const float* uhyper  = (const float*)d_in[2];
    const float* ihyper  = (const float*)d_in[3];
    const float* vals    = (const float*)d_in[4];
    const float* hgnn_uW = (const float*)d_in[5];
    const float* hgnn_iW = (const float*)d_in[6];
    const float* gcn_uW  = (const float*)d_in[7];
    const float* gcn_iW  = (const float*)d_in[8];
    const int*   rows    = (const int*)d_in[9];
    const int*   cols    = (const int*)d_in[10];

    const int U  = in_sizes[0] / 128;
    const int I  = in_sizes[1] / 128;
    const int E  = in_sizes[4];
    const int NL = in_sizes[5] / (3 * 128 * 128);

    const size_t UO = (size_t)U * 128, IO = (size_t)I * 128;
    float* ws   = (float*)d_ws;
    float* uu   = ws;            // [U,128] hyper adjacency
    float* ii   = uu + UO;       // [I,128]
    float* ulat = ii + IO;       // [U,128]
    float* ilat = ulat + UO;     // [I,128]
    float* hU   = ilat + IO;     // [U,128]
    float* hI   = hU + UO;       // [I,128]
    float* tU   = hI + IO;       // [U,128] spmm accum / gU (in-place gemm)
    float* tI   = tU + UO;       // [I,128]
    float* lat128 = tI + IO;     // [128,128]

    float* uSum = (float*)d_out;
    float* iSum = uSum + UO;

    const dim3 blk(256);
    const int gU128 = (U + 127) / 128, gI128 = (I + 127) / 128;
    const int gUred = (U + RCHUNK - 1) / RCHUNK, gIred = (I + RCHUNK - 1) / RCHUNK;
    const long nU4 = (long)(UO / 4), nI4 = (long)(IO / 4);
    const int gUe = (int)((nU4 + 255) / 256), gIe = (int)((nI4 + 255) / 256);
    const int gEdge = (E + 7) / 8;

    // hyper projections
    gemm128_kernel<0><<<gU128, blk, 0, stream>>>(uEmbed0, uhyper, uu, U);
    gemm128_kernel<0><<<gI128, blk, 0, stream>>>(iEmbed0, ihyper, ii, I);
    // ulat = uSum = uEmbed0 ; ilat = iSum = iEmbed0
    copy2_kernel<<<gUe, blk, 0, stream>>>((const float4*)uEmbed0, (float4*)ulat, (float4*)uSum, nU4);
    copy2_kernel<<<gIe, blk, 0, stream>>>((const float4*)iEmbed0, (float4*)ilat, (float4*)iSum, nI4);

    for (int l = 0; l < NL; ++l) {
        // ---- user hgnn: hU = lrelu(uu @ res(lrelu(uu^T @ ulat))) ----
        zero_kernel<<<16, blk, 0, stream>>>((float4*)lat128, 4096);
        hyperT_kernel<<<gUred, blk, 0, stream>>>(uu, ulat, lat128, U);
        resblock_kernel<<<1, blk, 0, stream>>>(lat128, hgnn_uW + (size_t)l * 3 * 16384);
        gemm128_kernel<1><<<gU128, blk, 0, stream>>>(uu, lat128, hU, U);
        // ---- item hgnn ----
        zero_kernel<<<16, blk, 0, stream>>>((float4*)lat128, 4096);
        hyperT_kernel<<<gIred, blk, 0, stream>>>(ii, ilat, lat128, I);
        resblock_kernel<<<1, blk, 0, stream>>>(lat128, hgnn_iW + (size_t)l * 3 * 16384);
        gemm128_kernel<1><<<gI128, blk, 0, stream>>>(ii, lat128, hI, I);
        // ---- GCN: gU = relu(spmm_u(ilat) @ Wl) (lrelu∘relu == relu) ----
        zero_kernel<<<gUe, blk, 0, stream>>>((float4*)tU, nU4);
        spmm_kernel<<<gEdge, blk, 0, stream>>>(ilat, rows, cols, vals, tU, E);
        gemm128_kernel<2><<<gU128, blk, 0, stream>>>(tU, gcn_uW + (size_t)l * 16384, tU, U);
        zero_kernel<<<gIe, blk, 0, stream>>>((float4*)tI, nI4);
        spmm_kernel<<<gEdge, blk, 0, stream>>>(ulat, cols, rows, vals, tI, E);
        gemm128_kernel<2><<<gI128, blk, 0, stream>>>(tI, gcn_iW + (size_t)l * 16384, tI, I);
        // ---- combine: lat = g + h + lat ; sum += lat ----
        combine_kernel<<<gUe, blk, 0, stream>>>((const float4*)tU, (const float4*)hU,
                                                (float4*)ulat, (float4*)uSum, nU4);
        combine_kernel<<<gIe, blk, 0, stream>>>((const float4*)tI, (const float4*)hI,
                                                (float4*)ilat, (float4*)iSum, nI4);
    }
}
